// PointNetSetAbstraction_4080218931818
// MI455X (gfx1250) — matrix operand order repaired and checked by reading, not compiled, not run on hardware
//
#include <hip/hip_runtime.h>
#include <hip/hip_bf16.h>

#define B_   16
#define N_   4096
#define S_   1024
#define K_   32
#define D_   64
#define R2_  0.04f      // RADIUS^2
#define EPS_ 1e-5f

typedef __attribute__((ext_vector_type(16))) _Float16 v16h;
typedef __attribute__((ext_vector_type(8)))  _Float16 v8h;
typedef __attribute__((ext_vector_type(8)))  float    v8f;

// Workspace layout (bytes):
//   grp      : 0                 .. 2 MB      (16,1024,32) int32
//   wh       : 2 MB              .. +36864    f16 weights: W0h 64x96, W1h 64x64, W2h 128x64
//   bnparams : 2 MB + 36864      .. +2048     f32: sc0[64] sh0[64] sc1[64] sh1[64] sc2[128] sh2[128]
#define WS_GRP_BYTES   ((size_t)B_ * S_ * K_ * 4)
#define WH0_OFF        0
#define WH1_OFF        (64 * 96)
#define WH2_OFF        (64 * 96 + 64 * 64)
#define WH_TOTAL       (64 * 96 + 64 * 64 + 128 * 64)

// ---------------------------------------------------------------------------
// Kernel 0: fold weights -> f16 (zero-padded K) and BN params -> (scale, shift).
// ---------------------------------------------------------------------------
__global__ __launch_bounds__(256)
void wcvt_kernel(const float* __restrict__ W0, const float* __restrict__ W1,
                 const float* __restrict__ W2,
                 const float* __restrict__ b0, const float* __restrict__ g0,
                 const float* __restrict__ be0, const float* __restrict__ m0,
                 const float* __restrict__ v0,
                 const float* __restrict__ b1, const float* __restrict__ g1,
                 const float* __restrict__ be1, const float* __restrict__ m1,
                 const float* __restrict__ v1,
                 const float* __restrict__ b2, const float* __restrict__ g2,
                 const float* __restrict__ be2, const float* __restrict__ m2,
                 const float* __restrict__ v2,
                 _Float16* __restrict__ wh, float* __restrict__ bn)
{
    const int tid = blockIdx.x * 256 + threadIdx.x;
    const int nth = gridDim.x * 256;
    // W0: (64,67) -> (64,96) f16, zero padded
    for (int e = tid; e < 64 * 96; e += nth) {
        const int o = e / 96, c = e - o * 96;
        wh[WH0_OFF + e] = (c < 67) ? (_Float16)W0[o * 67 + c] : (_Float16)0.0f;
    }
    // W1: (64,64)
    for (int e = tid; e < 64 * 64; e += nth)
        wh[WH1_OFF + e] = (_Float16)W1[e];
    // W2: (128,64)
    for (int e = tid; e < 128 * 64; e += nth)
        wh[WH2_OFF + e] = (_Float16)W2[e];
    // BN folds: sc = g*rsqrt(v+eps); sh = (b-m)*sc + beta
    for (int o = tid; o < 64; o += nth) {
        const float sc = g0[o] * rsqrtf(v0[o] + EPS_);
        bn[o]      = sc;
        bn[64 + o] = (b0[o] - m0[o]) * sc + be0[o];
    }
    for (int o = tid; o < 64; o += nth) {
        const float sc = g1[o] * rsqrtf(v1[o] + EPS_);
        bn[128 + o] = sc;
        bn[192 + o] = (b1[o] - m1[o]) * sc + be1[o];
    }
    for (int o = tid; o < 128; o += nth) {
        const float sc = g2[o] * rsqrtf(v2[o] + EPS_);
        bn[256 + o] = sc;
        bn[384 + o] = (b2[o] - m2[o]) * sc + be2[o];
    }
}

// ---------------------------------------------------------------------------
// Kernel 1: furthest point sampling. One block per batch, xyz resident in LDS,
// per-thread distance registers, wave32 shuffle argmax + cross-wave reduce.
// ---------------------------------------------------------------------------
__global__ __launch_bounds__(1024)
void fps_kernel(const float* __restrict__ xyz, float* __restrict__ newxyz)
{
    __shared__ float sx[N_], sy[N_], sz[N_];
    __shared__ float wbv[32];
    __shared__ int   wbi[32];
    __shared__ int   sfar;

    const int b   = blockIdx.x;
    const int tid = threadIdx.x;
    const float* base = xyz + (size_t)b * N_ * 3;

    for (int p = tid; p < N_; p += 1024) {
        sx[p] = base[p * 3 + 0];
        sy[p] = base[p * 3 + 1];
        sz[p] = base[p * 3 + 2];
    }
    if (tid == 0) sfar = 0;
    __syncthreads();

    if (tid == 0) {
        newxyz[(size_t)(b * S_) * 3 + 0] = sx[0];
        newxyz[(size_t)(b * S_) * 3 + 1] = sy[0];
        newxyz[(size_t)(b * S_) * 3 + 2] = sz[0];
    }

    float dist[4];
#pragma unroll
    for (int t = 0; t < 4; ++t) dist[t] = 1e10f;

    const int wv = tid >> 5, lane = tid & 31;
    int far = 0;

    for (int i = 1; i < S_; ++i) {
        const float cx = sx[far], cy = sy[far], cz = sz[far];
        float bv = -1.0f; int bi = 0;
#pragma unroll
        for (int t = 0; t < 4; ++t) {
            const int p = tid + t * 1024;
            const float dx = sx[p] - cx, dy = sy[p] - cy, dz = sz[p] - cz;
            const float d  = dx * dx + dy * dy + dz * dz;
            const float nd = fminf(dist[t], d);
            dist[t] = nd;
            if (nd > bv || (nd == bv && p < bi)) { bv = nd; bi = p; }
        }
#pragma unroll
        for (int o = 16; o > 0; o >>= 1) {
            const float ov = __shfl_xor(bv, o, 32);
            const int   oi = __shfl_xor(bi, o, 32);
            if (ov > bv || (ov == bv && oi < bi)) { bv = ov; bi = oi; }
        }
        if (lane == 0) { wbv[wv] = bv; wbi[wv] = bi; }
        __syncthreads();
        if (wv == 0) {
            float v2 = wbv[lane];
            int   i2 = wbi[lane];
#pragma unroll
            for (int o = 16; o > 0; o >>= 1) {
                const float ov = __shfl_xor(v2, o, 32);
                const int   oi = __shfl_xor(i2, o, 32);
                if (ov > v2 || (ov == v2 && oi < i2)) { v2 = ov; i2 = oi; }
            }
            if (lane == 0) {
                sfar = i2;
                newxyz[(size_t)(b * S_ + i) * 3 + 0] = sx[i2];
                newxyz[(size_t)(b * S_ + i) * 3 + 1] = sy[i2];
                newxyz[(size_t)(b * S_ + i) * 3 + 2] = sz[i2];
            }
        }
        __syncthreads();
        far = sfar;
    }
}

// ---------------------------------------------------------------------------
// Kernel 2: ball query. One block per batch (xyz in LDS), one thread per
// centroid; ordered early-exit scan collecting the first 32 in-ball indices,
// padded with the first hit.
// ---------------------------------------------------------------------------
__global__ __launch_bounds__(1024)
void ballq_kernel(const float* __restrict__ xyz,
                  const float* __restrict__ newxyz,
                  int* __restrict__ grp)
{
    __shared__ float sx[N_], sy[N_], sz[N_];
    const int b = blockIdx.x;
    const int s = threadIdx.x;
    const float* base = xyz + (size_t)b * N_ * 3;
    for (int p = s; p < N_; p += 1024) {
        sx[p] = base[p * 3 + 0];
        sy[p] = base[p * 3 + 1];
        sz[p] = base[p * 3 + 2];
    }
    __syncthreads();

    const float qx = newxyz[(size_t)(b * S_ + s) * 3 + 0];
    const float qy = newxyz[(size_t)(b * S_ + s) * 3 + 1];
    const float qz = newxyz[(size_t)(b * S_ + s) * 3 + 2];

    const size_t gb = (size_t)(b * S_ + s) * K_;
    int cnt = 0, first = 0;
    for (int n = 0; n < N_ && cnt < K_; ++n) {
        const float dx = sx[n] - qx, dy = sy[n] - qy, dz = sz[n] - qz;
        const float d2 = dx * dx + dy * dy + dz * dz;
        if (d2 <= R2_) {
            if (cnt == 0) first = n;
            grp[gb + cnt] = n;
            ++cnt;
        }
    }
    for (; cnt < K_; ++cnt) grp[gb + cnt] = first;
}

// ---------------------------------------------------------------------------
// Fused MLP layer on a 128-row LDS tile using v_wmma_f32_16x16x32_f16.
// Each wave owns one 16-row M-tile. A-fragments (one per K-step) preloaded as
// 2x ds_load_b128; B-fragments are two contiguous global_load_b128 of the
// pre-converted f16 weights (L2-resident). Fully unrolled via template ints.
// ---------------------------------------------------------------------------
template <int KSTEPS, int NTILES>
__device__ static inline void mlp_layer(
    const _Float16* __restrict__ bufIn, int ldIn,
    const _Float16* __restrict__ Wh, int ldW,
    const float* __restrict__ sc, const float* __restrict__ sh,
    _Float16* __restrict__ bufOut, int ldOut,
    int wave, int lane)
{
    const int  mrow0 = wave * 16;
    const int  lhalf = lane & 15;
    const bool hi    = lane >= 16;
    const int  arow  = mrow0 + lhalf;

    // Preload A fragments: lanes 0-15 -> K base 0/+16, lanes 16-31 -> +8/+24.
    v16h afr[KSTEPS];
#pragma unroll
    for (int ks = 0; ks < KSTEPS; ++ks) {
        const _Float16* ap = bufIn + (size_t)arow * ldIn + ks * 32 + (hi ? 8 : 0);
        const v8h lo  = *(const v8h*)(ap);
        const v8h hi8 = *(const v8h*)(ap + 16);
        afr[ks] = __builtin_shufflevector(lo, hi8, 0, 1, 2, 3, 4, 5, 6, 7,
                                          8, 9, 10, 11, 12, 13, 14, 15);
    }

#pragma unroll
    for (int nt = 0; nt < NTILES; ++nt) {
        const int o = nt * 16 + lhalf;            // output channel
        v8f acc = {};
#pragma unroll
        for (int ks = 0; ks < KSTEPS; ++ks) {
            // B fragment: two 8-f16 chunks from row o of Wh at K base 0/+16 (lanes 0-15), +8/+24 (lanes 16-31).
            const _Float16* wp = Wh + (size_t)o * ldW + ks * 32 + (hi ? 8 : 0);
            const v8h blo = *(const v8h*)(wp);
            const v8h bhi = *(const v8h*)(wp + 16);
            const v16h bfrag = __builtin_shufflevector(blo, bhi, 0, 1, 2, 3, 4, 5, 6, 7,
                                                       8, 9, 10, 11, 12, 13, 14, 15);
            acc = __builtin_amdgcn_wmma_f32_16x16x32_f16(
                false, afr[ks], false, bfrag, (short)0, acc, false, false);
        }
        // Epilogue: folded bias+BN, ReLU, f16 store.
        const float scv = sc[o];
        const float shv = sh[o];
        const int rbase = mrow0 + (hi ? 8 : 0);
#pragma unroll
        for (int r = 0; r < 8; ++r) {
            float y = acc[r] * scv + shv;
            y = y > 0.0f ? y : 0.0f;
            bufOut[(size_t)(rbase + r) * ldOut + o] = (_Float16)y;
        }
    }
}

// ---------------------------------------------------------------------------
// Kernel 3: gather + h0 build + 3 fused WMMA layers + max-pool over samples.
// One block = 4 centroids = 128 rows; 8 waves, one M-tile each.
// ---------------------------------------------------------------------------
__global__ __launch_bounds__(256)
void mlp_kernel(const float* __restrict__ xyz, const float* __restrict__ points,
                const float* __restrict__ newxyz, const int* __restrict__ grp,
                const _Float16* __restrict__ wh, const float* __restrict__ bn,
                float* __restrict__ out)
{
    __shared__ _Float16 bufA[128 * 96];   // h0 (K padded 67->96); reused for h2
    __shared__ _Float16 bufB[128 * 128];  // h1 (stride 64); reused for h3 (stride 128)
    __shared__ int      sidx[128];

    const int tid = threadIdx.x;
    const int b   = blockIdx.x >> 8;
    const int s0  = (blockIdx.x & 255) * 4;

    if (tid < 128) {
        const int c = tid >> 5, k = tid & 31;
        sidx[tid] = grp[(size_t)(b * S_ + s0 + c) * K_ + k];
    }
    __syncthreads();

    // Build h0 = [grouped_xyz - new_xyz (3) | grouped_points (64) | zero pad (29)]
    for (int e = tid; e < 128 * 96; e += 256) {
        const int row = e / 96, col = e - row * 96;
        const int c = row >> 5;
        const int i = sidx[row];
        float val = 0.0f;
        if (col < 3) {
            val = xyz[((size_t)b * N_ + i) * 3 + col]
                - newxyz[((size_t)b * S_ + s0 + c) * 3 + col];
        } else if (col < 67) {
            val = points[((size_t)b * N_ + i) * D_ + (col - 3)];
        }
        bufA[e] = (_Float16)val;
    }
    __syncthreads();

    const int wave = tid >> 5, lane = tid & 31;
    mlp_layer<3, 4>(bufA, 96, wh + WH0_OFF, 96, bn,       bn + 64,  bufB, 64,  wave, lane);
    __syncthreads();
    mlp_layer<2, 4>(bufB, 64, wh + WH1_OFF, 64, bn + 128, bn + 192, bufA, 64,  wave, lane);
    __syncthreads();
    mlp_layer<2, 8>(bufA, 64, wh + WH2_OFF, 64, bn + 256, bn + 384, bufB, 128, wave, lane);
    __syncthreads();

    // Max-pool over the 32 samples of each centroid.
    for (int vi = tid; vi < 512; vi += 256) {
        const int c = vi >> 7, ch = vi & 127;
        float mx = (float)bufB[(size_t)(c * 32) * 128 + ch];
        for (int k = 1; k < 32; ++k) {
            mx = fmaxf(mx, (float)bufB[(size_t)(c * 32 + k) * 128 + ch]);
        }
        out[((size_t)b * S_ + s0 + c) * 128 + ch] = mx;
    }
}

// ---------------------------------------------------------------------------
extern "C" void kernel_launch(void* const* d_in, const int* in_sizes, int n_in,
                              void* d_out, int out_size, void* d_ws, size_t ws_size,
                              hipStream_t stream)
{
    const float* xyz    = (const float*)d_in[0];
    const float* points = (const float*)d_in[1];
    const float* W0  = (const float*)d_in[2];
    const float* b0  = (const float*)d_in[3];
    const float* g0  = (const float*)d_in[4];
    const float* be0 = (const float*)d_in[5];
    const float* m0  = (const float*)d_in[6];
    const float* v0  = (const float*)d_in[7];
    const float* W1  = (const float*)d_in[8];
    const float* b1  = (const float*)d_in[9];
    const float* g1  = (const float*)d_in[10];
    const float* be1 = (const float*)d_in[11];
    const float* m1  = (const float*)d_in[12];
    const float* v1  = (const float*)d_in[13];
    const float* W2  = (const float*)d_in[14];
    const float* b2  = (const float*)d_in[15];
    const float* g2  = (const float*)d_in[16];
    const float* be2 = (const float*)d_in[17];
    const float* m2  = (const float*)d_in[18];
    const float* v2  = (const float*)d_in[19];

    float*    out    = (float*)d_out;
    float*    newxyz = out;                        // (16,1024,3)  = 49152 floats
    float*    newpts = out + (size_t)B_ * S_ * 3;  // (16,1024,128)
    int*      grp    = (int*)d_ws;                 // (16,1024,32) int32 = 2 MB
    _Float16* wh     = (_Float16*)((char*)d_ws + WS_GRP_BYTES);
    float*    bn     = (float*)((char*)d_ws + WS_GRP_BYTES + (size_t)WH_TOTAL * 2);

    wcvt_kernel <<<16, 256, 0, stream>>>(W0, W1, W2,
                                         b0, g0, be0, m0, v0,
                                         b1, g1, be1, m1, v1,
                                         b2, g2, be2, m2, v2, wh, bn);
    fps_kernel  <<<B_, 1024, 0, stream>>>(xyz, newxyz);
    ballq_kernel<<<B_, 1024, 0, stream>>>(xyz, newxyz, grp);
    mlp_kernel  <<<B_ * (S_ / 4), 256, 0, stream>>>(xyz, points, newxyz, grp,
                                                    wh, bn, newpts);
}
